// RefinerTransformer_77601469104648
// MI455X (gfx1250) — matrix operand order repaired and checked by reading, not compiled, not run on hardware
//
#include <hip/hip_runtime.h>
#include <hip/hip_bf16.h>
#include <stdint.h>

// ---------------------------------------------------------------------------
// Problem constants (from the reference)
// ---------------------------------------------------------------------------
#define BB   8
#define NN   2048
#define CC   128
#define KNBR 16     // final neighbors kept
#define NSEL 32     // K*DIL candidates extracted (sorted), keep even ranks

typedef __attribute__((ext_vector_type(16))) __bf16 v16bf;
typedef __attribute__((ext_vector_type(8)))  __bf16 v8bf;
typedef __attribute__((ext_vector_type(8)))  float  v8f;
typedef __attribute__((ext_vector_type(4)))  int    v4i;

#if defined(__has_builtin)
#if __has_builtin(__builtin_amdgcn_global_load_async_to_lds_b128)
#define USE_ASYNC_LDS 1
#endif
#endif

// ---------------------------------------------------------------------------
// WMMA helpers (CDNA5 16x16x32 bf16, fp32 accumulate)
// ---------------------------------------------------------------------------
__device__ __forceinline__ v8f wmma_bf16(v16bf a, v16bf b, v8f c) {
    // (neg_a, A, neg_b, B, c_mod, C, reuse_a, reuse_b)
    return __builtin_amdgcn_wmma_f32_16x16x32_bf16(false, a, false, b,
                                                   (short)0, c, false, false);
}

// A fragment: 16x32 (MxK) bf16, row-major source with leading dim `ld`.
// Lane layout: lanes 0-15 -> K 0..15, lanes 16-31 -> K 16..31 (contiguous).
__device__ __forceinline__ v16bf load_a_frag(const __bf16* base, int ld) {
    const int lane = threadIdx.x & 31;
    const __bf16* r = base + (size_t)(lane & 15) * ld + 16 * (lane >> 4);
    v8bf c0 = *(const v8bf*)(r);
    v8bf c1 = *(const v8bf*)(r + 8);
    v16bf f;
#pragma unroll
    for (int e = 0; e < 8; ++e) { f[e] = c0[e]; f[8 + e] = c1[e]; }
    return f;
}

// B fragment: 32x16 (KxN) bf16 from K-transposed storage Bt[n*ld + k]
// (each lane reads 16 contiguous bf16 = 32 bytes).
__device__ __forceinline__ v16bf load_b_frag(const __bf16* base, int ld) {
    const int lane = threadIdx.x & 31;
    const __bf16* r = base + (size_t)(lane & 15) * ld + 16 * (lane >> 4);
    return *(const v16bf*)r;
}

__device__ __forceinline__ void wait_async_all() {
#if defined(__has_builtin) && __has_builtin(__builtin_amdgcn_s_wait_asynccnt)
    __builtin_amdgcn_s_wait_asynccnt(0);
#else
    asm volatile("s_wait_asynccnt 0" ::: "memory");
#endif
}

// Stage 64 KB (hi||lo transposed weight block) into LDS.
// Async path: GLOBAL_LOAD_ASYNC_TO_LDS_B128 (ASYNCcnt), 16 B per lane per op.
__device__ __forceinline__ void stage_weights_lds(const __bf16* __restrict__ wsrc,
                                                  __bf16* sdst) {
#ifdef USE_ASYNC_LDS
    typedef __attribute__((address_space(1))) v4i* gp_t;
    typedef __attribute__((address_space(3))) v4i* lp_t;
    gp_t g = (gp_t)wsrc;     // C-style cast: const + addrspace conversion
    lp_t l = (lp_t)sdst;
#pragma unroll
    for (int it = 0; it < 32; ++it) {
        const int chunk = it * 128 + (int)threadIdx.x;   // 16-byte units
        __builtin_amdgcn_global_load_async_to_lds_b128(g + chunk, l + chunk, 0, 0);
    }
    wait_async_all();
    __syncthreads();
#else
#pragma unroll
    for (int it = 0; it < 32; ++it) {
        const int off = (it * 128 + (int)threadIdx.x) * 16;
        *(float4*)((char*)sdst + off) = *(const float4*)((const char*)wsrc + off);
    }
    __syncthreads();
#endif
}

// ---------------------------------------------------------------------------
// Kernel 1: split x -> bf16 hi/lo, row squared norms
// ---------------------------------------------------------------------------
__global__ __launch_bounds__(128)
void prep_x_kernel(const float* __restrict__ x, __bf16* __restrict__ xhi,
                   __bf16* __restrict__ xlo, float* __restrict__ sq) {
    const size_t row = blockIdx.x;
    const int c = threadIdx.x;
    float v = x[row * CC + c];
    __bf16 h = (__bf16)v;
    xhi[row * CC + c] = h;
    xlo[row * CC + c] = (__bf16)(v - (float)h);
    float s = v * v;
#pragma unroll
    for (int off = 16; off > 0; off >>= 1) s += __shfl_xor(s, off, 32);
    __shared__ float part[4];
    if ((threadIdx.x & 31) == 0) part[threadIdx.x >> 5] = s;
    __syncthreads();
    if (threadIdx.x == 0) sq[row] = part[0] + part[1] + part[2] + part[3];
}

// ---------------------------------------------------------------------------
// Kernel 2: per-batch channel max of x (global max pool)
// ---------------------------------------------------------------------------
__global__ __launch_bounds__(128)
void xmax_kernel(const float* __restrict__ x, float* __restrict__ xmax) {
    const int b = blockIdx.x, c = threadIdx.x;
    float m = -3.4e38f;
    for (int n = 0; n < NN; ++n)
        m = fmaxf(m, x[((size_t)b * NN + n) * CC + c]);
    xmax[b * CC + c] = m;
}

// ---------------------------------------------------------------------------
// Kernel 3: transpose + bf16 hi/lo split of the 4 128x128 weight blocks
// (W_src, W_dst, W_val, W_mlp[:128]); out: per-matrix [hi 16384][lo 16384]
// with layout Wt[n*128 + k] = W[k*128 + n].
// ---------------------------------------------------------------------------
__global__ __launch_bounds__(256)
void prep_w_kernel(const float* __restrict__ Wsrc, const float* __restrict__ Wdst,
                   const float* __restrict__ Wval, const float* __restrict__ Wmlp,
                   __bf16* __restrict__ o) {
    const float* srcs[4] = {Wsrc, Wdst, Wval, Wmlp};
    const int m = blockIdx.y;
    const int t = blockIdx.x * 256 + threadIdx.x;   // 0..16383
    const int n = t >> 7, k = t & 127;
    float v = srcs[m][(size_t)k * CC + n];
    __bf16 h = (__bf16)v;
    o[(size_t)m * 32768 + t] = h;
    o[(size_t)m * 32768 + 16384 + t] = (__bf16)(v - (float)h);
}

// ---------------------------------------------------------------------------
// Kernel 4: h2term[b, :] = xmax[b] @ W_mlp[128:256, :] + b_mlp   (tiny, fp32)
// ---------------------------------------------------------------------------
__global__ __launch_bounds__(128)
void h2term_kernel(const float* __restrict__ xmax, const float* __restrict__ Wmlp,
                   const float* __restrict__ bmlp, float* __restrict__ h2t) {
    const int b = blockIdx.x, co = threadIdx.x;
    float s = bmlp[co];
#pragma unroll 4
    for (int ci = 0; ci < CC; ++ci)
        s += xmax[b * CC + ci] * Wmlp[(size_t)(CC + ci) * CC + co];
    h2t[b * CC + co] = s;
}

// ---------------------------------------------------------------------------
// Kernel 5: the three projections  a_src/a_dst/a_val = x @ W  (WMMA, split bf16)
// Weight block staged in LDS via async copy; block = 4 waves, wave w owns
// cols [32w, 32w+32); grid = (rows/16, 3). Dynamic LDS: 64 KB.
// ---------------------------------------------------------------------------
__global__ __launch_bounds__(128)
void gemm3_kernel(const __bf16* __restrict__ xhi, const __bf16* __restrict__ xlo,
                  const __bf16* __restrict__ wts, float* __restrict__ asrc,
                  float* __restrict__ adst, float* __restrict__ aval) {
    extern __shared__ float dynsm[];
    __bf16* swt = (__bf16*)dynsm;                  // [hi 16384][lo 16384]
    const int i0 = blockIdx.x * 16;
    const int wsel = blockIdx.y;
    float* outp = (wsel == 0) ? asrc : (wsel == 1) ? adst : aval;

    stage_weights_lds(wts + (size_t)wsel * 32768, swt);
    const __bf16* wth = swt;
    const __bf16* wtl = swt + 16384;

    const int wv = threadIdx.x >> 5;
    const int n0 = wv * 32;
    const __bf16* arh = xhi + (size_t)i0 * CC;
    const __bf16* arl = xlo + (size_t)i0 * CC;

    v8f acc0 = {}, acc1 = {};
#pragma unroll
    for (int ks = 0; ks < 4; ++ks) {
        v16bf ah = load_a_frag(arh + ks * 32, CC);
        v16bf al = load_a_frag(arl + ks * 32, CC);
        v16bf bh0 = load_b_frag(wth + (size_t)n0 * CC + ks * 32, CC);
        v16bf bl0 = load_b_frag(wtl + (size_t)n0 * CC + ks * 32, CC);
        v16bf bh1 = load_b_frag(wth + (size_t)(n0 + 16) * CC + ks * 32, CC);
        v16bf bl1 = load_b_frag(wtl + (size_t)(n0 + 16) * CC + ks * 32, CC);
        acc0 = wmma_bf16(ah, bh0, acc0);
        acc0 = wmma_bf16(ah, bl0, acc0);
        acc0 = wmma_bf16(al, bh0, acc0);
        acc1 = wmma_bf16(ah, bh1, acc1);
        acc1 = wmma_bf16(ah, bl1, acc1);
        acc1 = wmma_bf16(al, bh1, acc1);
    }
    const int lane = threadIdx.x & 31;
    const int n = lane & 15, hp = lane >> 4;
#pragma unroll
    for (int r = 0; r < 8; ++r) {
        const int m = r + 8 * hp;
        outp[(size_t)(i0 + m) * CC + n0 + n]      = acc0[r];
        outp[(size_t)(i0 + m) * CC + n0 + 16 + n] = acc1[r];
    }
}

// ---------------------------------------------------------------------------
// Kernel 6: fused KNN — WMMA Gram strip (16 rows x 2048 cols) into LDS,
// then sorted argmin extraction with the per-lane candidates held in
// registers (64 floats/lane, static indices only; removal via bitmask).
// Dynamic LDS: 16*2048*4 = 128 KB (WGP has 320 KB).
// ---------------------------------------------------------------------------
__global__ __launch_bounds__(128)
void knn_kernel(const __bf16* __restrict__ xhi, const __bf16* __restrict__ xlo,
                const float* __restrict__ sq, int* __restrict__ idxout) {
    extern __shared__ float dynsm[];
    float* sd2 = dynsm;                            // [16][2048]
    const int tile = blockIdx.x;                   // B * N/16 tiles
    const int b = tile >> 7;                       // N/16 == 128
    const int i0 = (tile & 127) * 16;
    const int wv = threadIdx.x >> 5;
    const int lane = threadIdx.x & 31;
    const int n = lane & 15, hp = lane >> 4;

    const __bf16* xbh = xhi + (size_t)b * NN * CC;
    const __bf16* xbl = xlo + (size_t)b * NN * CC;
    const float* sqb = sq + (size_t)b * NN;

    // Preload the A strip (16 rows, full K=128) in registers.
    v16bf ah[4], al[4];
#pragma unroll
    for (int ks = 0; ks < 4; ++ks) {
        ah[ks] = load_a_frag(xbh + (size_t)i0 * CC + ks * 32, CC);
        al[ks] = load_a_frag(xbl + (size_t)i0 * CC + ks * 32, CC);
    }
    float sqi[8];
#pragma unroll
    for (int r = 0; r < 8; ++r) sqi[r] = sqb[i0 + r + 8 * hp];

    // Distance strip via split-bf16 WMMA: d2 = |xi|^2 + |xj|^2 - 2 xi.xj
    for (int t = wv; t < NN / 16; t += 4) {
        const int j0 = t * 16;
        if (t + 4 < NN / 16)
            __builtin_prefetch(xbh + (size_t)(j0 + 64) * CC, 0, 1);
        v8f acc = {};
#pragma unroll
        for (int ks = 0; ks < 4; ++ks) {
            v16bf bh = load_b_frag(xbh + (size_t)j0 * CC + ks * 32, CC);
            v16bf bl = load_b_frag(xbl + (size_t)j0 * CC + ks * 32, CC);
            acc = wmma_bf16(ah[ks], bh, acc);
            acc = wmma_bf16(ah[ks], bl, acc);
            acc = wmma_bf16(al[ks], bh, acc);
        }
        const int j = j0 + n;
        const float sqj = sqb[j];
#pragma unroll
        for (int r = 0; r < 8; ++r) {
            const int m = r + 8 * hp;
            float d = sqi[r] + sqj - 2.0f * acc[r];
            if (i0 + m == j) d = 1e10f;            // exclude self (eye*1e10)
            sd2[m * NN + j] = d;
        }
    }
    __syncthreads();

    // Sorted top-32 extraction; each wave owns 4 rows; lane owns 64 entries,
    // loaded once from LDS into registers.
    for (int rr = 0; rr < 4; ++rr) {
        const int m = wv * 4 + rr;
        const float* row = sd2 + (size_t)m * NN;
        float vals[64];
#pragma unroll
        for (int t2 = 0; t2 < 64; ++t2) vals[t2] = row[lane + (t2 << 5)];
        unsigned long long removed = 0ull;
        for (int r = 0; r < NSEL; ++r) {
            float best = 3.4e38f;
            int bidx = 1 << 30;
#pragma unroll
            for (int t2 = 0; t2 < 64; ++t2) {       // static indices only
                const float v = ((removed >> t2) & 1ull) ? 3.4e38f : vals[t2];
                const int j = lane + (t2 << 5);
                if (v < best || (v == best && j < bidx)) { best = v; bidx = j; }
            }
#pragma unroll
            for (int off = 16; off > 0; off >>= 1) {   // argmin, ties -> low idx
                const float ov = __shfl_xor(best, off, 32);
                const int oi = __shfl_xor(bidx, off, 32);
                if (ov < best || (ov == best && oi < bidx)) { best = ov; bidx = oi; }
            }
            if ((r & 1) == 0 && lane == 0)
                idxout[((size_t)(b * NN + i0 + m)) * KNBR + (r >> 1)] = bidx;
            if ((bidx & 31) == lane) removed |= 1ull << (bidx >> 5);
        }
    }
}

// ---------------------------------------------------------------------------
// Kernel 7: gather + positional delta + per-channel softmax over K, h1 (bf16 split)
// ---------------------------------------------------------------------------
__global__ __launch_bounds__(128)
void attn_kernel(const float* __restrict__ adst, const float* __restrict__ asrc,
                 const float* __restrict__ aval, const float* __restrict__ pos,
                 const float* __restrict__ Wpos, const float* __restrict__ bpos,
                 const int* __restrict__ idx, __bf16* __restrict__ h1hi,
                 __bf16* __restrict__ h1lo) {
    const size_t node = blockIdx.x;                // b*N + i
    const int b = (int)(node >> 11);
    const int c = threadIdx.x;
    __shared__ int sidx[KNBR];
    __shared__ float spi[3];
    if (threadIdx.x < KNBR) sidx[threadIdx.x] = idx[node * KNBR + threadIdx.x];
    if (threadIdx.x < 3)    spi[threadIdx.x] = pos[node * 3 + threadIdx.x];
    __syncthreads();

    const float w0 = Wpos[c], w1 = Wpos[CC + c], w2 = Wpos[2 * CC + c];
    const float bp = bpos[c];
    const float ad = adst[node * CC + c];

    float logit[KNBR], val[KNBR];
    float mx = -3.4e38f;
#pragma unroll
    for (int kk = 0; kk < KNBR; ++kk) {
        const size_t nj = (size_t)b * NN + sidx[kk];
        const float d0 = spi[0] - pos[nj * 3 + 0];
        const float d1 = spi[1] - pos[nj * 3 + 1];
        const float d2 = spi[2] - pos[nj * 3 + 2];
        const float delta = d0 * w0 + d1 * w1 + d2 * w2 + bp;
        const float l = ad - asrc[nj * CC + c] + delta;
        logit[kk] = l;
        val[kk] = aval[nj * CC + c] + delta;
        mx = fmaxf(mx, l);
    }
    float se = 0.0f, h = 0.0f;
#pragma unroll
    for (int kk = 0; kk < KNBR; ++kk) {
        const float e = expf(logit[kk] - mx);
        se += e;
        h += e * val[kk];
    }
    h /= se;
    const __bf16 hh = (__bf16)h;
    h1hi[node * CC + c] = hh;
    h1lo[node * CC + c] = (__bf16)(h - (float)hh);
}

// ---------------------------------------------------------------------------
// Kernel 8: out = relu(h1 @ W_mlp[:128] + h2term[b])   (WMMA, split bf16)
// Weight block staged in LDS via async copy. Dynamic LDS: 64 KB.
// ---------------------------------------------------------------------------
__global__ __launch_bounds__(128)
void mlp_kernel(const __bf16* __restrict__ h1hi, const __bf16* __restrict__ h1lo,
                const __bf16* __restrict__ wts, const float* __restrict__ h2t,
                float* __restrict__ out) {
    extern __shared__ float dynsm[];
    __bf16* swt = (__bf16*)dynsm;
    const int i0 = blockIdx.x * 16;
    const int b = i0 >> 11;

    stage_weights_lds(wts + (size_t)3 * 32768, swt);  // transposed W_mlp[:128]
    const __bf16* wth = swt;
    const __bf16* wtl = swt + 16384;

    const int wv = threadIdx.x >> 5;
    const int n0 = wv * 32;

    v8f acc0 = {}, acc1 = {};
#pragma unroll
    for (int ks = 0; ks < 4; ++ks) {
        v16bf ahv = load_a_frag(h1hi + (size_t)i0 * CC + ks * 32, CC);
        v16bf alv = load_a_frag(h1lo + (size_t)i0 * CC + ks * 32, CC);
        v16bf bh0 = load_b_frag(wth + (size_t)n0 * CC + ks * 32, CC);
        v16bf bl0 = load_b_frag(wtl + (size_t)n0 * CC + ks * 32, CC);
        v16bf bh1 = load_b_frag(wth + (size_t)(n0 + 16) * CC + ks * 32, CC);
        v16bf bl1 = load_b_frag(wtl + (size_t)(n0 + 16) * CC + ks * 32, CC);
        acc0 = wmma_bf16(ahv, bh0, acc0);
        acc0 = wmma_bf16(ahv, bl0, acc0);
        acc0 = wmma_bf16(alv, bh0, acc0);
        acc1 = wmma_bf16(ahv, bh1, acc1);
        acc1 = wmma_bf16(ahv, bl1, acc1);
        acc1 = wmma_bf16(alv, bh1, acc1);
    }
    const int lane = threadIdx.x & 31;
    const int n = lane & 15, hp = lane >> 4;
#pragma unroll
    for (int r = 0; r < 8; ++r) {
        const int m = r + 8 * hp;
        const float v0 = acc0[r] + h2t[b * CC + n0 + n];
        const float v1 = acc1[r] + h2t[b * CC + n0 + 16 + n];
        out[(size_t)(i0 + m) * CC + n0 + n]      = fmaxf(v0, 0.0f);
        out[(size_t)(i0 + m) * CC + n0 + 16 + n] = fmaxf(v1, 0.0f);
    }
}

// ---------------------------------------------------------------------------
// Host launcher
// ---------------------------------------------------------------------------
extern "C" void kernel_launch(void* const* d_in, const int* in_sizes, int n_in,
                              void* d_out, int out_size, void* d_ws, size_t ws_size,
                              hipStream_t stream) {
    (void)in_sizes; (void)n_in; (void)out_size; (void)ws_size;
    const float* x    = (const float*)d_in[0];
    const float* pos  = (const float*)d_in[1];
    const float* Wsrc = (const float*)d_in[2];
    const float* Wdst = (const float*)d_in[3];
    const float* Wval = (const float*)d_in[4];
    const float* Wpos = (const float*)d_in[5];
    const float* bpos = (const float*)d_in[6];
    const float* Wmlp = (const float*)d_in[7];
    const float* bmlp = (const float*)d_in[8];
    float* out = (float*)d_out;

    const size_t rows = (size_t)BB * NN;           // 16384
    const size_t xel  = rows * CC;                 // 2,097,152

    char* p = (char*)d_ws;
    auto take = [&](size_t bytes) -> char* {
        char* r = p;
        p += (bytes + 255) & ~(size_t)255;
        return r;
    };
    __bf16* xhi  = (__bf16*)take(xel * 2);
    __bf16* xlo  = (__bf16*)take(xel * 2);
    float*  sq   = (float*)take(rows * 4);
    float*  xmax = (float*)take((size_t)BB * CC * 4);
    __bf16* wts  = (__bf16*)take((size_t)4 * 32768 * 2);
    float*  h2t  = (float*)take((size_t)BB * CC * 4);
    float*  asrc = (float*)take(xel * 4);
    float*  adst = (float*)take(xel * 4);
    float*  aval = (float*)take(xel * 4);
    int*    idx  = (int*)take(rows * KNBR * 4);
    __bf16* h1hi = (__bf16*)take(xel * 2);
    __bf16* h1lo = (__bf16*)take(xel * 2);

    prep_x_kernel<<<(int)rows, CC, 0, stream>>>(x, xhi, xlo, sq);
    xmax_kernel<<<BB, CC, 0, stream>>>(x, xmax);
    prep_w_kernel<<<dim3(64, 4), 256, 0, stream>>>(Wsrc, Wdst, Wval, Wmlp, wts);
    h2term_kernel<<<BB, CC, 0, stream>>>(xmax, Wmlp, bmlp, h2t);

    // 64 KB dynamic LDS for the staged weight block
    (void)hipFuncSetAttribute((const void*)gemm3_kernel,
                              hipFuncAttributeMaxDynamicSharedMemorySize, 65536);
    gemm3_kernel<<<dim3((int)(rows / 16), 3), 128, 65536, stream>>>(
        xhi, xlo, wts, asrc, adst, aval);

    // 128 KB dynamic LDS for the 16x2048 distance strip (WGP has 320 KB)
    (void)hipFuncSetAttribute((const void*)knn_kernel,
                              hipFuncAttributeMaxDynamicSharedMemorySize, 16 * NN * 4);
    knn_kernel<<<(int)(rows / 16), 128, 16 * NN * 4, stream>>>(xhi, xlo, sq, idx);

    attn_kernel<<<(int)rows, CC, 0, stream>>>(adst, asrc, aval, pos, Wpos, bpos,
                                              idx, h1hi, h1lo);

    (void)hipFuncSetAttribute((const void*)mlp_kernel,
                              hipFuncAttributeMaxDynamicSharedMemorySize, 65536);
    mlp_kernel<<<(int)(rows / 16), 128, 65536, stream>>>(h1hi, h1lo, wts, h2t, out);
}
